// Pointnet2Seg_72791105733156
// MI455X (gfx1250) — matrix operand order repaired and checked by reading, not compiled, not run on hardware
//
#include <hip/hip_runtime.h>

// ---------------------------------------------------------------------------
// PointNet++ segmentation pipeline for MI455X (gfx1250, wave32, WMMA).
// All activations point-major: (B, Npts, C). All GEMM-shaped work uses
// v_wmma_f32_16x16x32_bf16 (bf16 A/B, f32 accum). LDS tiles are stored
// pre-swizzled in WMMA fragment order so fragment loads are ds_load_b128.
// ---------------------------------------------------------------------------

typedef __attribute__((ext_vector_type(16))) __bf16 v16bf;
typedef __attribute__((ext_vector_type(8)))  float  v8f;
typedef __attribute__((ext_vector_type(4)))  unsigned int uint4v;
typedef unsigned short ushort_t;

#define EPS_BN 1e-5f

union FragBF { ushort_t u[16]; unsigned d[8]; uint4v q[2]; v16bf v; };

// f32 -> bf16, round-half-up (2 VALU ops; bf16 ulp makes RNE vs RHU moot here)
__device__ __forceinline__ ushort_t f2bf(float f) {
  return (ushort_t)((__float_as_uint(f) + 0x8000u) >> 16);
}
__device__ __forceinline__ unsigned pack2bf(float a, float b) {
  return ((__float_as_uint(a) + 0x8000u) >> 16) |
         ((__float_as_uint(b) + 0x8000u) & 0xffff0000u);
}

// A fragment (16x32 bf16, M=lane%16, g=lane/16) from a 32-entry bf16 row in
// LDS: each lane takes K = [g*16, g*16+16) (contiguous order, matching B),
// i.e. one contiguous 32B run -> two ds_load_b128.
__device__ __forceinline__ v16bf a_frag_lds(const ushort_t* row, int g) {
  FragBF A;
  const uint4v* p = (const uint4v*)row;   // 16B-aligned (row stride 64B)
  A.q[0] = p[2 * g];
  A.q[1] = p[2 * g + 1];
  return A.v;
}
// Same fragment straight from a global f32 row (one 64B run via float4).
__device__ __forceinline__ v16bf a_frag_global(const float* row, int g) {
  FragBF A;
  const float4* p = (const float4*)row;
  float4 f0 = p[4 * g], f1 = p[4 * g + 1];
  float4 f2 = p[4 * g + 2], f3 = p[4 * g + 3];
  A.d[0] = pack2bf(f0.x, f0.y); A.d[1] = pack2bf(f0.z, f0.w);
  A.d[2] = pack2bf(f1.x, f1.y); A.d[3] = pack2bf(f1.z, f1.w);
  A.d[4] = pack2bf(f2.x, f2.y); A.d[5] = pack2bf(f2.z, f2.w);
  A.d[6] = pack2bf(f3.x, f3.y); A.d[7] = pack2bf(f3.z, f3.w);
  return A.v;
}
// B fragment from fragment-ordered LDS: 16 contiguous bf16 (32B) per lane.
__device__ __forceinline__ v16bf b_frag_vec(const ushort_t* p) {
  FragBF B;
  const uint4v* q = (const uint4v*)p;     // 32B-aligned
  B.q[0] = q[0];
  B.q[1] = q[1];
  return B.v;
}

// ---------------------------------------------------------------------------
// 1) Embedding: per point, 3->64 and 64->64 pointwise convs with BN+ReLU.
// ---------------------------------------------------------------------------
__global__ void __launch_bounds__(128) embed_kernel(
    const float* __restrict__ x,           // B,N,3
    const float* __restrict__ w1,          // 64x3
    const float* __restrict__ g1, const float* __restrict__ be1,
    const float* __restrict__ w2,          // 64x64
    const float* __restrict__ g2, const float* __restrict__ be2,
    float* __restrict__ out,               // B,N,64
    int N) {
  __shared__ float sw2[64 * 64];
  __shared__ float sw1[64 * 3], ss1[64], sb1[64], ss2[64], sb2[64];
  int tid = threadIdx.x;
  for (int e = tid; e < 64 * 64; e += 128) sw2[e] = w2[e];
  for (int e = tid; e < 64 * 3; e += 128) sw1[e] = w1[e];
  const float sc = rsqrtf(1.f + EPS_BN);
  for (int e = tid; e < 64; e += 128) {
    ss1[e] = g1[e] * sc; sb1[e] = be1[e];
    ss2[e] = g2[e] * sc; sb2[e] = be2[e];
  }
  __syncthreads();
  int n = blockIdx.x * 128 + tid;
  int b = blockIdx.y;
  if (n >= N) return;
  const float* p = x + ((size_t)b * N + n) * 3;
  float px = p[0], py = p[1], pz = p[2];
  float h1[64];
#pragma unroll 8
  for (int o = 0; o < 64; ++o) {
    float a = sw1[o * 3] * px + sw1[o * 3 + 1] * py + sw1[o * 3 + 2] * pz;
    h1[o] = fmaxf(a * ss1[o] + sb1[o], 0.f);
  }
  float* op = out + ((size_t)b * N + n) * 64;
  for (int o = 0; o < 64; ++o) {
    float a = 0.f;
#pragma unroll 8
    for (int i = 0; i < 64; ++i) a += sw2[o * 64 + i] * h1[i];
    op[o] = fmaxf(a * ss2[o] + sb2[o], 0.f);
  }
}

// ---------------------------------------------------------------------------
// 2) Farthest point sampling: one block per batch, dist[] in LDS.
// ---------------------------------------------------------------------------
__global__ void __launch_bounds__(1024) fps_kernel(
    const float* __restrict__ xyz, int* __restrict__ idx, int N, int npoint) {
  extern __shared__ float dist[];         // N floats
  __shared__ float rv[1024];
  __shared__ int ri[1024];
  __shared__ int s_far;
  int b = blockIdx.x, tid = threadIdx.x, nt = blockDim.x;
  const float* xb = xyz + (size_t)b * N * 3;
  for (int i = tid; i < N; i += nt) dist[i] = 1e10f;
  if (tid == 0) s_far = 0;
  __syncthreads();
  for (int j = 0; j < npoint; ++j) {
    int far = s_far;
    if (tid == 0) idx[b * npoint + j] = far;
    float cx = xb[3 * far], cy = xb[3 * far + 1], cz = xb[3 * far + 2];
    float bv = -1.f; int bi = 0;
    for (int i = tid; i < N; i += nt) {
      float dx = xb[3 * i] - cx, dy = xb[3 * i + 1] - cy, dz = xb[3 * i + 2] - cz;
      float m = fminf(dist[i], dx * dx + dy * dy + dz * dz);
      dist[i] = m;
      if (m > bv) { bv = m; bi = i; }
    }
    rv[tid] = bv; ri[tid] = bi;
    __syncthreads();
    for (int off = nt >> 1; off > 0; off >>= 1) {
      if (tid < off) {
        bool take = (rv[tid + off] > rv[tid]) ||
                    (rv[tid + off] == rv[tid] && ri[tid + off] < ri[tid]);
        if (take) { rv[tid] = rv[tid + off]; ri[tid] = ri[tid + off]; }
      }
      __syncthreads();
    }
    if (tid == 0) s_far = ri[0];
    __syncthreads();
  }
}

__global__ void gather_xyz_kernel(const float* __restrict__ xyz,
                                  const int* __restrict__ idx,
                                  float* __restrict__ out, int N, int S) {
  int t = blockIdx.x * blockDim.x + threadIdx.x;
  int b = blockIdx.y;
  if (t >= S * 3) return;
  int s = t / 3, c = t % 3;
  out[((size_t)b * S + s) * 3 + c] = xyz[((size_t)b * N + idx[b * S + s]) * 3 + c];
}

// ---------------------------------------------------------------------------
// 3) Ball query: one wave32 per query; ballot + prefix-popcount compaction
//    gives the first 32 in-range indices in ascending order.
// ---------------------------------------------------------------------------
__global__ void __launch_bounds__(256) query_ball_kernel(
    const float* __restrict__ xyz, const float* __restrict__ new_xyz,
    int* __restrict__ ball, int N, int S, float r2) {
  int lane = threadIdx.x & 31;
  int s = blockIdx.x * 8 + (threadIdx.x >> 5);
  int b = blockIdx.y;
  if (s >= S) return;
  const float* xb = xyz + (size_t)b * N * 3;
  const float* q = new_xyz + ((size_t)b * S + s) * 3;
  float qx = q[0], qy = q[1], qz = q[2];
  int* bp = ball + ((size_t)b * S + s) * 32;
  int cnt = 0, firstj = -1;
  for (int base = 0; base < N && cnt < 32; base += 32) {
    int j = base + lane;
    bool pred = false;
    if (j < N) {
      float dx = xb[3 * j] - qx, dy = xb[3 * j + 1] - qy, dz = xb[3 * j + 2] - qz;
      pred = (dx * dx + dy * dy + dz * dz) <= r2;
    }
    unsigned mask = (unsigned)__ballot(pred);
    if (firstj < 0 && mask) firstj = base + __builtin_ctz(mask);
    int pos = cnt + __popc(mask & ((1u << lane) - 1u));
    if (pred && pos < 32) bp[pos] = j;
    cnt += __popc(mask);
  }
  if (firstj < 0) firstj = 0;
  for (int t = cnt + lane; t < 32; t += 32) bp[t] = firstj;
}

// ---------------------------------------------------------------------------
// 4) local_op: per group of 32 samples, two WMMA GEMM stages + channel max.
//    C = prev feature dim, CO = output dim; Cin = 2C (centered | replicated).
//    X and H tiles live in LDS in WMMA B-fragment order:
//    F[k32][h][lane][t] = M[k32*32 + (lane/16)*16 + t][h*16 + lane%16]
// ---------------------------------------------------------------------------
template <int C, int CO>
__global__ void __launch_bounds__(256) local_op_kernel(
    const float* __restrict__ feats_prev,  // B,Nprev,C
    const int* __restrict__ fps_idx,       // B,S
    const int* __restrict__ ball_idx,      // B,S,32
    const float* __restrict__ w1, const float* __restrict__ g1, const float* __restrict__ b1,
    const float* __restrict__ w2, const float* __restrict__ g2, const float* __restrict__ b2,
    float* __restrict__ out,               // B,S,CO
    int Nprev, int S) {
  constexpr int CI = 2 * C;
  __shared__ alignas(16) ushort_t XsF[CI / 32][2][32][16];
  __shared__ alignas(16) ushort_t HsF[CO / 32][2][32][16];
  __shared__ float s_new[C];
  __shared__ int s_ball[32];
  int s = blockIdx.x, b = blockIdx.y;
  int tid = threadIdx.x, lane = tid & 31, wave = tid >> 5;
  const float* fb = feats_prev + (size_t)b * Nprev * C;

  if (tid < 32) s_ball[tid] = ball_idx[((size_t)b * S + s) * 32 + tid];
  int ci = fps_idx[b * S + s];
  for (int c = tid; c < C; c += 256) s_new[c] = fb[(size_t)ci * C + c];
  __syncthreads();

  // build X in fragment order; each thread handles one run of 8 channels
  constexpr int RPK = CI / 8;              // runs per sample
  for (int rix = tid; rix < 32 * RPK; rix += 256) {
    int k = rix / RPK;                     // sample 0..31
    int i0 = (rix % RPK) * 8;              // channel run start
    float f[8];
    if (i0 < C) {
      const float* src = fb + (size_t)s_ball[k] * C + i0;
      float4 a = *(const float4*)src;
      float4 c4 = *(const float4*)(src + 4);
      f[0] = a.x - s_new[i0];      f[1] = a.y - s_new[i0 + 1];
      f[2] = a.z - s_new[i0 + 2];  f[3] = a.w - s_new[i0 + 3];
      f[4] = c4.x - s_new[i0 + 4]; f[5] = c4.y - s_new[i0 + 5];
      f[6] = c4.z - s_new[i0 + 6]; f[7] = c4.w - s_new[i0 + 7];
    } else {
#pragma unroll
      for (int t = 0; t < 8; ++t) f[t] = s_new[i0 - C + t];
    }
    uint4v pk;
    pk[0] = pack2bf(f[0], f[1]); pk[1] = pack2bf(f[2], f[3]);
    pk[2] = pack2bf(f[4], f[5]); pk[3] = pack2bf(f[6], f[7]);
    int gk = (i0 >> 4) & 1, t0 = i0 & 15;
    *(uint4v*)&XsF[i0 >> 5][k >> 4][gk * 16 + (k & 15)][t0] = pk;
  }
  __syncthreads();

  const float scb = rsqrtf(1.f + EPS_BN);
  int g = lane >> 4, l15 = lane & 15;

  // ---- GEMM1: H = relu(bn(W1 (COxCI) * X (CIx32))) -> HsF (bf16) ----
  for (int mt = wave; mt < CO / 16; mt += 8) {
    v8f a0 = {}, a1 = {};
    const float* wrow = w1 + (size_t)(mt * 16 + l15) * CI;
    for (int kc = 0; kc < CI; kc += 32) {
      if (kc + 32 < CI) __builtin_prefetch(wrow + kc + 32, 0, 1);
      v16bf A  = a_frag_global(wrow + kc, g);
      v16bf B0 = b_frag_vec(XsF[kc >> 5][0][lane]);
      v16bf B1 = b_frag_vec(XsF[kc >> 5][1][lane]);
      a0 = __builtin_amdgcn_wmma_f32_16x16x32_bf16(false, A, false, B0, (short)0, a0, false, false);
      a1 = __builtin_amdgcn_wmma_f32_16x16x32_bf16(false, A, false, B1, (short)0, a1, false, false);
    }
#pragma unroll
    for (int r = 0; r < 8; ++r) {
      int m = mt * 16 + r + 8 * g;
      float sc = g1[m] * scb, sh = b1[m];
      int gk = (m >> 4) & 1, tt = m & 15, L = gk * 16 + l15;
      HsF[m >> 5][0][L][tt] = f2bf(fmaxf(a0[r] * sc + sh, 0.f));
      HsF[m >> 5][1][L][tt] = f2bf(fmaxf(a1[r] * sc + sh, 0.f));
    }
  }
  __syncthreads();

  // ---- GEMM2 + BN + ReLU + max over samples, write out[b,s,m] ----
  for (int mt = wave; mt < CO / 16; mt += 8) {
    v8f a0 = {}, a1 = {};
    const float* wrow = w2 + (size_t)(mt * 16 + l15) * CO;
    for (int kc = 0; kc < CO; kc += 32) {
      if (kc + 32 < CO) __builtin_prefetch(wrow + kc + 32, 0, 1);
      v16bf A  = a_frag_global(wrow + kc, g);
      v16bf B0 = b_frag_vec(HsF[kc >> 5][0][lane]);
      v16bf B1 = b_frag_vec(HsF[kc >> 5][1][lane]);
      a0 = __builtin_amdgcn_wmma_f32_16x16x32_bf16(false, A, false, B0, (short)0, a0, false, false);
      a1 = __builtin_amdgcn_wmma_f32_16x16x32_bf16(false, A, false, B1, (short)0, a1, false, false);
    }
#pragma unroll
    for (int r = 0; r < 8; ++r) {
      int m = mt * 16 + r + 8 * g;
      float sc = g2[m] * scb, sh = b2[m];
      float vm = fmaxf(fmaxf(a0[r] * sc + sh, 0.f), fmaxf(a1[r] * sc + sh, 0.f));
#pragma unroll
      for (int off = 1; off < 16; off <<= 1) vm = fmaxf(vm, __shfl_xor(vm, off, 32));
      if (l15 == 0) out[((size_t)b * S + s) * CO + m] = vm;
    }
  }
}

// ---------------------------------------------------------------------------
// 5) 3-NN + inverse-distance weights (feature propagation).
// ---------------------------------------------------------------------------
__global__ void knn3_kernel(const float* __restrict__ xyz1,  // B,S1,3
                            const float* __restrict__ xyz2,  // B,S2,3
                            int* __restrict__ oidx, float* __restrict__ ow,
                            int S1, int S2) {
  int n = blockIdx.x * blockDim.x + threadIdx.x;
  int b = blockIdx.y;
  if (n >= S1) return;
  const float* p1 = xyz1 + ((size_t)b * S1 + n) * 3;
  float px = p1[0], py = p1[1], pz = p1[2];
  float d0 = 1e30f, d1 = 1e30f, d2 = 1e30f;
  int i0 = 0, i1 = 0, i2 = 0;
  const float* p2 = xyz2 + (size_t)b * S2 * 3;
  for (int j = 0; j < S2; ++j) {
    float dx = px - p2[3 * j], dy = py - p2[3 * j + 1], dz = pz - p2[3 * j + 2];
    float d = dx * dx + dy * dy + dz * dz;
    if (d < d0)      { d2 = d1; i2 = i1; d1 = d0; i1 = i0; d0 = d; i0 = j; }
    else if (d < d1) { d2 = d1; i2 = i1; d1 = d; i1 = j; }
    else if (d < d2) { d2 = d; i2 = j; }
  }
  float r0 = 1.f / (d0 + 1e-8f), r1 = 1.f / (d1 + 1e-8f), r2 = 1.f / (d2 + 1e-8f);
  float rs = 1.f / (r0 + r1 + r2);
  size_t o = ((size_t)b * S1 + n) * 3;
  oidx[o] = i0; oidx[o + 1] = i1; oidx[o + 2] = i2;
  ow[o] = r0 * rs; ow[o + 1] = r1 * rs; ow[o + 2] = r2 * rs;
}

__global__ void concat_interp_kernel(const float* __restrict__ p1,  // B,N,C1
                                     const float* __restrict__ f2,  // B,S2,C2
                                     const int* __restrict__ idx3,
                                     const float* __restrict__ w3,
                                     float* __restrict__ cat,       // B,N,C1+C2
                                     int N, int C1, int C2, int S2) {
  int t = blockIdx.x * blockDim.x + threadIdx.x;
  int b = blockIdx.y;
  int Ct = C1 + C2;
  if (t >= N * Ct) return;
  int n = t / Ct, c = t % Ct;
  float v;
  if (c < C1) {
    v = p1[((size_t)b * N + n) * C1 + c];
  } else {
    int cc = c - C1;
    const int* id = idx3 + ((size_t)b * N + n) * 3;
    const float* w = w3 + ((size_t)b * N + n) * 3;
    const float* fb = f2 + (size_t)b * S2 * C2;
    v = w[0] * fb[(size_t)id[0] * C2 + cc] +
        w[1] * fb[(size_t)id[1] * C2 + cc] +
        w[2] * fb[(size_t)id[2] * C2 + cc];
  }
  cat[((size_t)b * N + n) * Ct + c] = v;
}

// ---------------------------------------------------------------------------
// 6) Generic WMMA GEMM: Y[b,n,m] = epi( sum_k W[m,k] * X[b,n,k] )
//    epi = relu? max(0, (acc + bias[m]) * (gamma[m]/sqrt(1+eps)) + beta[m])
//    Block: 128 threads (4 waves), tile 64(M) x 32(N), K in chunks of 32.
//    W tile row-major bf16 in LDS (A loads = 2x ds_load_b128/lane),
//    X tile fragment-ordered bf16 in LDS (B loads = 2x ds_load_b128/lane).
// ---------------------------------------------------------------------------
__global__ void __launch_bounds__(128) gemm_bn_relu_kernel(
    const float* __restrict__ W, const float* __restrict__ X, float* __restrict__ Y,
    const float* __restrict__ gamma, const float* __restrict__ beta,
    const float* __restrict__ bias, int M, int K, int Npts, int relu) {
  __shared__ alignas(16) ushort_t Ws[64][32];
  __shared__ alignas(16) ushort_t XsF[2][32][16];
  int m0 = blockIdx.x * 64;
  int n0 = blockIdx.y * 32;
  int b = blockIdx.z;
  const float* Xb = X + (size_t)b * Npts * K;
  float* Yb = Y + (size_t)b * Npts * M;
  int tid = threadIdx.x, lane = tid & 31, wave = tid >> 5;
  int g = lane >> 4, l15 = lane & 15;
  // staging coords: W -> row r, 16-col half; X -> point nn, 8-k run
  int wr = tid >> 1, wh = (tid & 1) * 16;
  int xn = tid >> 2, xk0 = (tid & 3) * 8;
  v8f acc0 = {}, acc1 = {};
  for (int kc = 0; kc < K; kc += 32) {
    {  // W tile: 64x32, each thread loads 16 contiguous f32 -> 32B LDS store
      int m = m0 + wr;
      uint4v p0, p1;
      if (m < M) {
        const float4* src = (const float4*)(W + (size_t)m * K + kc + wh);
        float4 a = src[0], c4 = src[1], e4 = src[2], h4 = src[3];
        p0[0] = pack2bf(a.x, a.y);   p0[1] = pack2bf(a.z, a.w);
        p0[2] = pack2bf(c4.x, c4.y); p0[3] = pack2bf(c4.z, c4.w);
        p1[0] = pack2bf(e4.x, e4.y); p1[1] = pack2bf(e4.z, e4.w);
        p1[2] = pack2bf(h4.x, h4.y); p1[3] = pack2bf(h4.z, h4.w);
      } else {
        p0 = (uint4v)0; p1 = (uint4v)0;
      }
      *(uint4v*)&Ws[wr][wh] = p0;
      *(uint4v*)&Ws[wr][wh + 8] = p1;
    }
    {  // X tile: 32 points x 32 k, fragment-ordered
      const float4* src = (const float4*)(Xb + (size_t)(n0 + xn) * K + kc + xk0);
      float4 a = src[0], c4 = src[1];
      uint4v pk;
      pk[0] = pack2bf(a.x, a.y);   pk[1] = pack2bf(a.z, a.w);
      pk[2] = pack2bf(c4.x, c4.y); pk[3] = pack2bf(c4.z, c4.w);
      *(uint4v*)&XsF[xn >> 4][((xk0 >> 4) & 1) * 16 + (xn & 15)][xk0 & 15] = pk;
    }
    __syncthreads();
    v16bf A  = a_frag_lds(Ws[wave * 16 + l15], g);
    v16bf B0 = b_frag_vec(XsF[0][lane]);
    v16bf B1 = b_frag_vec(XsF[1][lane]);
    acc0 = __builtin_amdgcn_wmma_f32_16x16x32_bf16(false, A, false, B0, (short)0, acc0, false, false);
    acc1 = __builtin_amdgcn_wmma_f32_16x16x32_bf16(false, A, false, B1, (short)0, acc1, false, false);
    __syncthreads();
  }
  const float scb = rsqrtf(1.f + EPS_BN);
  int mb = m0 + wave * 16 + 8 * g;   // 8 consecutive output channels per lane
  if (mb < M) {                      // M is always a multiple of 8
    float v0[8], v1[8];
#pragma unroll
    for (int r = 0; r < 8; ++r) {
      int m = mb + r;
      float sc = gamma ? gamma[m] * scb : 1.f;
      float sh = beta ? beta[m] : 0.f;
      float bi = bias ? bias[m] : 0.f;
      v0[r] = (acc0[r] + bi) * sc + sh;
      v1[r] = (acc1[r] + bi) * sc + sh;
      if (relu) { v0[r] = fmaxf(v0[r], 0.f); v1[r] = fmaxf(v1[r], 0.f); }
    }
    float4* y0 = (float4*)&Yb[(size_t)(n0 + l15) * M + mb];
    float4* y1 = (float4*)&Yb[(size_t)(n0 + 16 + l15) * M + mb];
    y0[0] = make_float4(v0[0], v0[1], v0[2], v0[3]);
    y0[1] = make_float4(v0[4], v0[5], v0[6], v0[7]);
    y1[0] = make_float4(v1[0], v1[1], v1[2], v1[3]);
    y1[1] = make_float4(v1[4], v1[5], v1[6], v1[7]);
  }
}

// ---------------------------------------------------------------------------
// Host orchestration
// ---------------------------------------------------------------------------
static void launch_gemm(const float* W, const float* X, float* Y,
                        const float* gamma, const float* beta, const float* bias,
                        int M, int K, int Npts, int relu, int B, hipStream_t st) {
  dim3 grid((M + 63) / 64, Npts / 32, B);
  gemm_bn_relu_kernel<<<grid, 128, 0, st>>>(W, X, Y, gamma, beta, bias, M, K, Npts, relu);
}

extern "C" void kernel_launch(void* const* d_in, const int* in_sizes, int n_in,
                              void* d_out, int out_size, void* d_ws, size_t ws_size,
                              hipStream_t stream) {
  (void)in_sizes; (void)n_in; (void)out_size; (void)ws_size;
  const int B = 4, N = 8192, S1 = 2048, S2 = 1024;
  const float R2 = 1.0f;  // radius^2

  const float* x      = (const float*)d_in[0];
  const float* w_emb1 = (const float*)d_in[1];
  const float* g_emb1 = (const float*)d_in[2];
  const float* be_emb1= (const float*)d_in[3];
  const float* w_emb2 = (const float*)d_in[4];
  const float* g_emb2 = (const float*)d_in[5];
  const float* be_emb2= (const float*)d_in[6];
  const float* l0_w1  = (const float*)d_in[7];
  const float* l0_g1  = (const float*)d_in[8];
  const float* l0_b1  = (const float*)d_in[9];
  const float* l0_w2  = (const float*)d_in[10];
  const float* l0_g2  = (const float*)d_in[11];
  const float* l0_b2  = (const float*)d_in[12];
  const float* l1_w1  = (const float*)d_in[13];
  const float* l1_g1  = (const float*)d_in[14];
  const float* l1_b1  = (const float*)d_in[15];
  const float* l1_w2  = (const float*)d_in[16];
  const float* l1_g2  = (const float*)d_in[17];
  const float* l1_b2  = (const float*)d_in[18];
  const float* f0_w1  = (const float*)d_in[19];
  const float* f0_g1  = (const float*)d_in[20];
  const float* f0_b1  = (const float*)d_in[21];
  const float* f0_w2  = (const float*)d_in[22];
  const float* f0_g2  = (const float*)d_in[23];
  const float* f0_b2  = (const float*)d_in[24];
  const float* f1_w1  = (const float*)d_in[25];
  const float* f1_g1  = (const float*)d_in[26];
  const float* f1_b1  = (const float*)d_in[27];
  const float* f1_w2  = (const float*)d_in[28];
  const float* f1_g2  = (const float*)d_in[29];
  const float* f1_b2  = (const float*)d_in[30];
  const float* c1_w   = (const float*)d_in[31];
  const float* c1_bias= (const float*)d_in[32];
  const float* c1_g   = (const float*)d_in[33];
  const float* c1_beta= (const float*)d_in[34];
  const float* c2_w   = (const float*)d_in[35];
  const float* c2_bias= (const float*)d_in[36];
  const float* c2_g   = (const float*)d_in[37];
  const float* c2_beta= (const float*)d_in[38];
  const float* out_w  = (const float*)d_in[39];
  const float* out_bias=(const float*)d_in[40];

  // --- workspace bump allocator (256B aligned) ---
  size_t off = 0;
  char* base = (char*)d_ws;
  auto alloc = [&](size_t bytes) -> void* {
    void* p = base + off;
    off += (bytes + 255) & ~(size_t)255;
    return p;
  };
  float* feats0 = (float*)alloc((size_t)B * N * 64 * 4);
  int*   fps1   = (int*)  alloc((size_t)B * S1 * 4);
  float* xyz1   = (float*)alloc((size_t)B * S1 * 3 * 4);
  int*   ball1  = (int*)  alloc((size_t)B * S1 * 32 * 4);
  float* feats1 = (float*)alloc((size_t)B * S1 * 128 * 4);
  int*   fps2   = (int*)  alloc((size_t)B * S2 * 4);
  float* xyz2   = (float*)alloc((size_t)B * S2 * 3 * 4);
  int*   ball2  = (int*)  alloc((size_t)B * S2 * 32 * 4);
  float* feats2 = (float*)alloc((size_t)B * S2 * 256 * 4);
  int*   knn0i  = (int*)  alloc((size_t)B * S1 * 3 * 4);
  float* knn0w  = (float*)alloc((size_t)B * S1 * 3 * 4);
  float* cat0   = (float*)alloc((size_t)B * S1 * 384 * 4);
  float* fp0a   = (float*)alloc((size_t)B * S1 * 256 * 4);
  float* fp0b   = (float*)alloc((size_t)B * S1 * 512 * 4);
  int*   knn1i  = (int*)  alloc((size_t)B * N * 3 * 4);
  float* knn1w  = (float*)alloc((size_t)B * N * 3 * 4);
  float* cat1   = (float*)alloc((size_t)B * N * 576 * 4);
  float* fp1a   = (float*)alloc((size_t)B * N * 512 * 4);
  float* fp1b   = (float*)alloc((size_t)B * N * 1024 * 4);
  float* c1o = fp1a;   // fp1a dead after fp1b GEMM
  float* c2o = cat1;   // cat1 dead after fp1a GEMM

  // 1) embedding -> feats0 (B,N,64)
  embed_kernel<<<dim3((N + 127) / 128, B), 128, 0, stream>>>(
      x, w_emb1, g_emb1, be_emb1, w_emb2, g_emb2, be_emb2, feats0, N);

  // 2) FPS stage 1 (8192 -> 2048) + gather + ball query
  fps_kernel<<<B, 1024, N * sizeof(float), stream>>>(x, fps1, N, S1);
  gather_xyz_kernel<<<dim3((S1 * 3 + 255) / 256, B), 256, 0, stream>>>(x, fps1, xyz1, N, S1);
  query_ball_kernel<<<dim3(S1 / 8, B), 256, 0, stream>>>(x, xyz1, ball1, N, S1, R2);

  // 3) local_op stage 1: (64 -> 128)
  local_op_kernel<64, 128><<<dim3(S1, B), 256, 0, stream>>>(
      feats0, fps1, ball1, l0_w1, l0_g1, l0_b1, l0_w2, l0_g2, l0_b2, feats1, N, S1);

  // 4) FPS stage 2 (2048 -> 1024) + gather + ball query
  fps_kernel<<<B, 1024, S1 * sizeof(float), stream>>>(xyz1, fps2, S1, S2);
  gather_xyz_kernel<<<dim3((S2 * 3 + 255) / 256, B), 256, 0, stream>>>(xyz1, fps2, xyz2, S1, S2);
  query_ball_kernel<<<dim3(S2 / 8, B), 256, 0, stream>>>(xyz1, xyz2, ball2, S1, S2, R2);

  // 5) local_op stage 2: (128 -> 256)
  local_op_kernel<128, 256><<<dim3(S2, B), 256, 0, stream>>>(
      feats1, fps2, ball2, l1_w1, l1_g1, l1_b1, l1_w2, l1_g2, l1_b2, feats2, S1, S2);

  // 6) feature_prop 0: interp feats2 @ xyz1, concat with feats1 -> 384 -> 256 -> 512
  knn3_kernel<<<dim3(S1 / 256, B), 256, 0, stream>>>(xyz1, xyz2, knn0i, knn0w, S1, S2);
  concat_interp_kernel<<<dim3((S1 * 384 + 255) / 256, B), 256, 0, stream>>>(
      feats1, feats2, knn0i, knn0w, cat0, S1, 128, 256, S2);
  launch_gemm(f0_w1, cat0, fp0a, f0_g1, f0_b1, nullptr, 256, 384, S1, 1, B, stream);
  launch_gemm(f0_w2, fp0a, fp0b, f0_g2, f0_b2, nullptr, 512, 256, S1, 1, B, stream);

  // 7) feature_prop 1: interp fp0b @ xyz, concat with feats0 -> 576 -> 512 -> 1024
  knn3_kernel<<<dim3(N / 256, B), 256, 0, stream>>>(x, xyz1, knn1i, knn1w, N, S1);
  concat_interp_kernel<<<dim3((N * 576 + 255) / 256, B), 256, 0, stream>>>(
      feats0, fp0b, knn1i, knn1w, cat1, N, 64, 512, S1);
  launch_gemm(f1_w1, cat1, fp1a, f1_g1, f1_b1, nullptr, 512, 576, N, 1, B, stream);
  launch_gemm(f1_w2, fp1a, fp1b, f1_g2, f1_b2, nullptr, 1024, 512, N, 1, B, stream);

  // 8) head: c1 (bias+bn+relu), c2 (bias+bn+relu), out (bias only) -> d_out (B,N,8)
  launch_gemm(c1_w, fp1b, c1o, c1_g, c1_beta, c1_bias, 512, 1024, N, 1, B, stream);
  launch_gemm(c2_w, c1o, c2o, c2_g, c2_beta, c2_bias, 128, 512, N, 1, B, stream);
  launch_gemm(out_w, c2o, (float*)d_out, nullptr, nullptr, out_bias, 8, 128, N, 0, B, stream);
}
